// RoleConditionedLSTMDecoder_24008867184973
// MI455X (gfx1250) — matrix operand order repaired and checked by reading, not compiled, not run on hardware
//
#include <hip/hip_runtime.h>

typedef __attribute__((ext_vector_type(16))) __bf16 v16bf;
typedef __attribute__((ext_vector_type(8)))  __bf16 v8bf;
typedef __attribute__((ext_vector_type(8)))  float  v8f;

// Problem constants
constexpr int Bn  = 1024;
constexpr int Pn  = 22;
constexpr int Nn  = Bn * Pn;        // 22528
constexpr int CDn = 256;
constexpr int RDn = 64;
constexpr int Hn  = 512;
constexpr int G4  = 4 * Hn;         // 2048
constexpr int INn = 324;            // [pos2, ball2, ctx256, role64]
constexpr int KPn = 352;            // padded K for static GEMM (mult of 32)
constexpr int HHn = 256;            // H/2
constexpr int ODn = 2;
constexpr int ROWS = 16;            // rows per workgroup tile
constexpr int NWG  = Nn / ROWS;     // 1408

// Workspace layout (bytes)
constexpr size_t OFF_WIH0P = 0;                                    // [2048][352] bf16
constexpr size_t SZ_WIH0P  = (size_t)G4 * KPn * 2;
constexpr size_t OFF_WHH0  = OFF_WIH0P + SZ_WIH0P;                 // [2048][512] bf16
constexpr size_t SZ_W512   = (size_t)G4 * Hn * 2;
constexpr size_t OFF_WIH1  = OFF_WHH0 + SZ_W512;
constexpr size_t OFF_WHH1  = OFF_WIH1 + SZ_W512;
constexpr size_t OFF_WFC1  = OFF_WHH1 + SZ_W512;                   // [256][512] bf16
constexpr size_t SZ_WFC1   = (size_t)HHn * Hn * 2;
constexpr size_t OFF_SG    = OFF_WFC1 + SZ_WFC1;                   // [N][2048] bf16

// Dynamic LDS layout for decoder kernel (bytes)
constexpr int L_SG   = 0;                       // 16*2048 bf16 = 65536
constexpr int L_HA0  = L_SG + ROWS * G4 * 2;    // 2 buffers * 16*512 bf16 = 32768
constexpr int L_HA1  = L_HA0 + 2 * ROWS * Hn * 2;
constexpr int L_HID  = L_HA1 + 2 * ROWS * Hn * 2;   // 16*256 bf16 = 8192
constexpr int L_WFC2 = L_HID + ROWS * HHn * 2;      // 512 f32 = 2048
constexpr int L_POS  = L_WFC2 + ODn * HHn * 4;      // 32 f32
constexpr int L_RED  = L_POS + ROWS * ODn * 4;      // 256 f32 = 1024
constexpr int SMEM_BYTES = L_RED + 256 * 4;         // ~142.5 KB

__device__ inline float sigmoidf_(float x) { return 1.0f / (1.0f + __expf(-x)); }

// A fragment: 16x32 bf16, lane m = lane&15 holds row m.
// lanes 0-15: K = k0+0..7 and k0+16..23 ; lanes 16-31: K = k0+8..15 and k0+24..31
__device__ inline v16bf load_a_frag(const __bf16* __restrict__ base, int stride, int k0, int lane) {
    int m    = lane & 15;
    int half = lane >> 4;
    const __bf16* p = base + (size_t)m * stride + k0 + half * 8;
    v8bf lo = *(const v8bf*)(p);
    v8bf hi = *(const v8bf*)(p + 16);
    v16bf a;
#pragma unroll
    for (int i = 0; i < 8; ++i) { a[i] = lo[i]; a[i + 8] = hi[i]; }
    return a;
}

// B fragment: 32x16 (KxN) from row-major W[out][in]; lane n = lane&15 is col n.
__device__ inline v16bf load_b_frag(const __bf16* __restrict__ W, int stride, int n0, int k0, int lane) {
    int nn   = lane & 15;
    int half = lane >> 4;
    const __bf16* p = W + (size_t)(n0 + nn) * stride + k0 + half * 8;
    v8bf lo = *(const v8bf*)(p);
    v8bf hi = *(const v8bf*)(p + 16);
    v16bf b;
#pragma unroll
    for (int i = 0; i < 8; ++i) { b[i] = lo[i]; b[i + 8] = hi[i]; }
    return b;
}

__device__ inline v8f wmma_bf16(v16bf a, v16bf b, v8f c) {
    return __builtin_amdgcn_wmma_f32_16x16x32_bf16(false, a, false, b, (short)0, c, false, false);
}

// CDNA5 async global->LDS copy (16B), tracked by ASYNCcnt
__device__ inline void async_copy_b128(unsigned lds_byte_addr, const void* gptr) {
    asm volatile("global_load_async_to_lds_b128 %0, %1, off"
                 :: "v"(lds_byte_addr), "v"(gptr) : "memory");
}

// ---------------------------------------------------------------------------
// Kernel 1: convert weights to bf16 (W_ih0 zero-padded to K=352)
// ---------------------------------------------------------------------------
__global__ void convert_weights_kernel(const float* __restrict__ Wih0,
                                       const float* __restrict__ Whh0,
                                       const float* __restrict__ Wih1,
                                       const float* __restrict__ Whh1,
                                       const float* __restrict__ Wfc1,
                                       __bf16* __restrict__ o_ih0p,
                                       __bf16* __restrict__ o_hh0,
                                       __bf16* __restrict__ o_ih1,
                                       __bf16* __restrict__ o_hh1,
                                       __bf16* __restrict__ o_fc1) {
    const long S0 = (long)G4 * KPn;
    const long S1 = (long)G4 * Hn;
    const long S4 = (long)HHn * Hn;
    const long total = S0 + 3 * S1 + S4;
    for (long i = blockIdx.x * (long)blockDim.x + threadIdx.x; i < total;
         i += (long)gridDim.x * blockDim.x) {
        if (i < S0) {
            long g = i / KPn, k = i % KPn;
            o_ih0p[i] = (k < INn) ? (__bf16)Wih0[g * INn + k] : (__bf16)0.0f;
        } else if (i < S0 + S1) {
            long j = i - S0;            o_hh0[j] = (__bf16)Whh0[j];
        } else if (i < S0 + 2 * S1) {
            long j = i - S0 - S1;       o_ih1[j] = (__bf16)Wih1[j];
        } else if (i < S0 + 3 * S1) {
            long j = i - S0 - 2 * S1;   o_hh1[j] = (__bf16)Whh1[j];
        } else {
            long j = i - S0 - 3 * S1;   o_fc1[j] = (__bf16)Wfc1[j];
        }
    }
}

// ---------------------------------------------------------------------------
// Kernel 2: static_gates[n][2048] = b0 + [0,0,ball,ctx,role,pad] @ W_ih0p^T
// ---------------------------------------------------------------------------
__global__ void __launch_bounds__(256)
static_gates_kernel(const float* __restrict__ context,
                    const float* __restrict__ ball,
                    const int*   __restrict__ roles,
                    const float* __restrict__ role_table,
                    const float* __restrict__ b0,
                    const __bf16* __restrict__ Wih0p,
                    __bf16* __restrict__ sg) {
    __shared__ __attribute__((aligned(32))) __bf16 As[ROWS * KPn];
    const int tid  = threadIdx.x;
    const int lane = tid & 31;
    const int wave = tid >> 5;
    const int n0   = blockIdx.x * ROWS;

    for (int i = tid; i < ROWS * KPn; i += blockDim.x) {
        int r = i / KPn, k = i % KPn;
        int n = n0 + r;
        float v = 0.0f;
        if (k >= 2 && k < 4)          v = ball[(n / Pn) * 2 + (k - 2)];
        else if (k >= 4 && k < 260)   v = context[(size_t)n * CDn + (k - 4)];
        else if (k >= 260 && k < INn) v = role_table[roles[n] * RDn + (k - 260)];
        As[i] = (__bf16)v;
    }
    __syncthreads();

    for (int cb = 0; cb < 16; ++cb) {
        const int col0 = (wave * 16 + cb) * 16;
        const float bv = b0[col0 + (lane & 15)];
        v8f acc;
#pragma unroll
        for (int r = 0; r < 8; ++r) acc[r] = bv;
        for (int kb = 0; kb < KPn / 32; ++kb) {
            v16bf a = load_a_frag(As, KPn, kb * 32, lane);
            v16bf b = load_b_frag(Wih0p, KPn, col0, kb * 32, lane);
            acc = wmma_bf16(a, b, acc);
        }
#pragma unroll
        for (int r = 0; r < 8; ++r) {
            int m   = r + 8 * (lane >> 4);
            int col = col0 + (lane & 15);
            sg[(size_t)(n0 + m) * G4 + col] = (__bf16)acc[r];
        }
    }
}

// ---------------------------------------------------------------------------
// Kernel 3: persistent per-row-tile decoder — all T steps in one launch.
// 8 wave32 waves; wave w owns hidden cols [w*64, w*64+64).
// c0/c1 cell state lives in registers in WMMA accumulator layout.
// h tiles double-buffered in LDS; static gates staged in LDS via async copy.
// ---------------------------------------------------------------------------
__global__ void __launch_bounds__(256)
lstm_decoder_kernel(float* __restrict__ out,
                    const __bf16* __restrict__ sg,
                    const float*  __restrict__ Wih0,    // f32, pos columns
                    const __bf16* __restrict__ Whh0b,
                    const __bf16* __restrict__ Wih1b,
                    const __bf16* __restrict__ Whh1b,
                    const __bf16* __restrict__ Wfc1b,
                    const float* __restrict__ b1,
                    const float* __restrict__ b_fc1,
                    const float* __restrict__ W_fc2,
                    const float* __restrict__ b_fc2,
                    const float* __restrict__ W_init,
                    const float* __restrict__ b_init,
                    const float* __restrict__ init_pos,
                    const int*   __restrict__ max_steps_p) {
    extern __shared__ __attribute__((aligned(32))) char smem[];
    __bf16* sgS   = (__bf16*)(smem + L_SG);    // [16][2048]
    __bf16* hA0b  = (__bf16*)(smem + L_HA0);   // 2 x [16][512]
    __bf16* hA1b  = (__bf16*)(smem + L_HA1);   // 2 x [16][512]
    __bf16* hid   = (__bf16*)(smem + L_HID);   // [16][256]
    float*  wfc2S = (float*)(smem + L_WFC2);   // [2][256]
    float*  posS  = (float*)(smem + L_POS);    // [16][2]
    float*  red   = (float*)(smem + L_RED);    // [256]

    const int tid  = threadIdx.x;
    const int lane = tid & 31;
    const int wave = tid >> 5;
    const int n0   = blockIdx.x * ROWS;
    const int T_steps = *max_steps_p;
    const int half8 = 8 * (lane >> 4);
    const int lcol  = lane & 15;

    // ---- async stage of per-tile static gates (64 KB), once for all T steps
    {
        const char* src = (const char*)(sg + (size_t)n0 * G4);
#pragma unroll
        for (int i = 0; i < 16; ++i) {
            int byte_off = (tid + i * 256) * 16;
            unsigned lds_addr = (unsigned)(uintptr_t)((char*)sgS + byte_off);
            async_copy_b128(lds_addr, src + byte_off);
        }
    }

    // ---- init: h0 = pos0 @ W_init^T + b_init ; h1 = 0 ; c = 0 ; pos = pos0
    if (tid < ROWS * ODn) {
        posS[tid] = init_pos[(size_t)(n0 + (tid >> 1)) * ODn + (tid & 1)];
    }
    for (int i = tid; i < ODn * HHn; i += 256) wfc2S[i] = W_fc2[i];
    {
        const int r  = tid & 15;
        const int c0 = (tid >> 4) * 32;
        const float px = init_pos[(size_t)(n0 + r) * ODn + 0];
        const float py = init_pos[(size_t)(n0 + r) * ODn + 1];
        for (int j = 0; j < 32; ++j) {
            int c = c0 + j;
            float h = b_init[c] + px * W_init[c * 2 + 0] + py * W_init[c * 2 + 1];
            hA0b[r * Hn + c] = (__bf16)h;
            hA1b[r * Hn + c] = (__bf16)0.0f;
        }
    }

    // ---- hoist loop-invariant scalars: pos-column weights and layer-1 bias
    float w0a[16], w1a[16], b1r[16];
#pragma unroll
    for (int hb = 0; hb < 4; ++hb)
#pragma unroll
        for (int gt = 0; gt < 4; ++gt) {
            const int col = gt * Hn + wave * 64 + hb * 16 + lcol;
            w0a[hb * 4 + gt] = Wih0[(size_t)col * INn + 0];
            w1a[hb * 4 + gt] = Wih0[(size_t)col * INn + 1];
            b1r[hb * 4 + gt] = b1[col];
        }
    float bfc1r[2];
#pragma unroll
    for (int cb = 0; cb < 2; ++cb) bfc1r[cb] = b_fc1[wave * 32 + cb * 16 + lcol];

    v8f c0f[4], c1f[4];
#pragma unroll
    for (int hb = 0; hb < 4; ++hb)
#pragma unroll
        for (int r = 0; r < 8; ++r) { c0f[hb][r] = 0.0f; c1f[hb][r] = 0.0f; }

    asm volatile("s_wait_asynccnt 0x0" ::: "memory");
    __syncthreads();

    int cur = 0;
    for (int t = 0; t < T_steps; ++t) {
        const __bf16* h0r = hA0b + cur * (ROWS * Hn);
        __bf16*       h0w = hA0b + (cur ^ 1) * (ROWS * Hn);
        const __bf16* h1r = hA1b + cur * (ROWS * Hn);
        __bf16*       h1w = hA1b + (cur ^ 1) * (ROWS * Hn);

        // ===== layer 0: gates = sg + pos-part + h0 @ Whh0^T
        for (int hb = 0; hb < 4; ++hb) {
            const int colI = wave * 64 + hb * 16;
            v8f acc[4];
#pragma unroll
            for (int gt = 0; gt < 4; ++gt) {
                const int col = gt * Hn + colI + lcol;
                const float w0 = w0a[hb * 4 + gt], w1 = w1a[hb * 4 + gt];
#pragma unroll
                for (int r = 0; r < 8; ++r) {
                    const int m = r + half8;
                    acc[gt][r] = (float)sgS[m * G4 + col]
                               + posS[m * 2 + 0] * w0 + posS[m * 2 + 1] * w1;
                }
            }
            for (int kb = 0; kb < Hn / 32; ++kb) {
                v16bf a = load_a_frag(h0r, Hn, kb * 32, lane);
#pragma unroll
                for (int gt = 0; gt < 4; ++gt) {
                    v16bf b = load_b_frag(Whh0b, Hn, gt * Hn + colI, kb * 32, lane);
                    acc[gt] = wmma_bf16(a, b, acc[gt]);
                }
            }
#pragma unroll
            for (int r = 0; r < 8; ++r) {
                float ig = sigmoidf_(acc[0][r]);
                float fg = sigmoidf_(acc[1][r]);
                float gg = tanhf(acc[2][r]);
                float og = sigmoidf_(acc[3][r]);
                float cn = fg * c0f[hb][r] + ig * gg;
                c0f[hb][r] = cn;
                h0w[(r + half8) * Hn + colI + lcol] = (__bf16)(og * tanhf(cn));
            }
        }
        __syncthreads();    // new h0 complete & old-h readers done

        // ===== layer 1: gates = b1 + h0new @ Wih1^T + h1 @ Whh1^T
        for (int hb = 0; hb < 4; ++hb) {
            const int colI = wave * 64 + hb * 16;
            v8f acc[4];
#pragma unroll
            for (int gt = 0; gt < 4; ++gt) {
                const float bv = b1r[hb * 4 + gt];
#pragma unroll
                for (int r = 0; r < 8; ++r) acc[gt][r] = bv;
            }
            for (int kb = 0; kb < Hn / 32; ++kb) {
                v16bf a = load_a_frag(h0w, Hn, kb * 32, lane);
#pragma unroll
                for (int gt = 0; gt < 4; ++gt) {
                    v16bf b = load_b_frag(Wih1b, Hn, gt * Hn + colI, kb * 32, lane);
                    acc[gt] = wmma_bf16(a, b, acc[gt]);
                }
            }
            for (int kb = 0; kb < Hn / 32; ++kb) {
                v16bf a = load_a_frag(h1r, Hn, kb * 32, lane);
#pragma unroll
                for (int gt = 0; gt < 4; ++gt) {
                    v16bf b = load_b_frag(Whh1b, Hn, gt * Hn + colI, kb * 32, lane);
                    acc[gt] = wmma_bf16(a, b, acc[gt]);
                }
            }
#pragma unroll
            for (int r = 0; r < 8; ++r) {
                float ig = sigmoidf_(acc[0][r]);
                float fg = sigmoidf_(acc[1][r]);
                float gg = tanhf(acc[2][r]);
                float og = sigmoidf_(acc[3][r]);
                float cn = fg * c1f[hb][r] + ig * gg;
                c1f[hb][r] = cn;
                h1w[(r + half8) * Hn + colI + lcol] = (__bf16)(og * tanhf(cn));
            }
        }
        __syncthreads();    // new h1 complete

        // ===== MLP fc1: hidden = relu(h1new @ Wfc1^T + b_fc1)
        for (int cb = 0; cb < 2; ++cb) {
            const int colI = wave * 32 + cb * 16;
            const float bv = bfc1r[cb];
            v8f acc;
#pragma unroll
            for (int r = 0; r < 8; ++r) acc[r] = bv;
            for (int kb = 0; kb < Hn / 32; ++kb) {
                v16bf a = load_a_frag(h1w, Hn, kb * 32, lane);
                v16bf b = load_b_frag(Wfc1b, Hn, colI, kb * 32, lane);
                acc = wmma_bf16(a, b, acc);
            }
#pragma unroll
            for (int r = 0; r < 8; ++r)
                hid[(r + half8) * HHn + colI + lcol] = (__bf16)fmaxf(acc[r], 0.0f);
        }
        __syncthreads();

        // ===== fc2: pos = hidden @ Wfc2^T + b_fc2 (8-way split dot + reduce)
        {
            const int g = tid >> 3, sub = tid & 7;
            const int row = g >> 1, od = g & 1;
            float p = 0.0f;
#pragma unroll
            for (int jj = 0; jj < 32; ++jj) {
                int j = sub * 32 + jj;
                p += (float)hid[row * HHn + j] * wfc2S[od * HHn + j];
            }
            red[tid] = p;
        }
        __syncthreads();
        if (tid < ROWS * ODn) {
            const int row = tid >> 1, od = tid & 1;
            float s = b_fc2[od];
#pragma unroll
            for (int k = 0; k < 8; ++k) s += red[tid * 8 + k];
            out[((size_t)(n0 + row) * T_steps + t) * ODn + od] = s;
            posS[tid] = s;
        }
        __syncthreads();

        cur ^= 1;
    }
}

// ---------------------------------------------------------------------------
extern "C" void kernel_launch(void* const* d_in, const int* in_sizes, int n_in,
                              void* d_out, int out_size, void* d_ws, size_t ws_size,
                              hipStream_t stream) {
    const float* context  = (const float*)d_in[0];
    const float* init_pos = (const float*)d_in[1];
    const float* ball     = (const float*)d_in[2];
    const int*   roles    = (const int*)  d_in[3];
    const int*   maxstep  = (const int*)  d_in[4];
    const float* roletab  = (const float*)d_in[5];
    const float* W_ih0    = (const float*)d_in[6];
    const float* W_hh0    = (const float*)d_in[7];
    const float* b0       = (const float*)d_in[8];
    const float* W_ih1    = (const float*)d_in[9];
    const float* W_hh1    = (const float*)d_in[10];
    const float* b1       = (const float*)d_in[11];
    const float* W_fc1    = (const float*)d_in[12];
    const float* b_fc1    = (const float*)d_in[13];
    const float* W_fc2    = (const float*)d_in[14];
    const float* b_fc2    = (const float*)d_in[15];
    const float* W_init   = (const float*)d_in[16];
    const float* b_init   = (const float*)d_in[17];
    float* out = (float*)d_out;

    char* ws = (char*)d_ws;
    __bf16* Wih0p = (__bf16*)(ws + OFF_WIH0P);
    __bf16* Whh0b = (__bf16*)(ws + OFF_WHH0);
    __bf16* Wih1b = (__bf16*)(ws + OFF_WIH1);
    __bf16* Whh1b = (__bf16*)(ws + OFF_WHH1);
    __bf16* Wfc1b = (__bf16*)(ws + OFF_WFC1);
    __bf16* sg    = (__bf16*)(ws + OFF_SG);

    // allow >64KB dynamic LDS for the decoder (one WG per WGP)
    (void)hipFuncSetAttribute((const void*)lstm_decoder_kernel,
                              hipFuncAttributeMaxDynamicSharedMemorySize,
                              SMEM_BYTES);

    // 1) weights -> bf16 (padded W_ih0)
    {
        long total = (long)G4 * KPn + 3L * G4 * Hn + (long)HHn * Hn;
        int blocks = (int)((total + 255) / 256);
        convert_weights_kernel<<<blocks, 256, 0, stream>>>(
            W_ih0, W_hh0, W_ih1, W_hh1, W_fc1, Wih0p, Whh0b, Wih1b, Whh1b, Wfc1b);
    }
    // 2) static input-gate contribution (once)
    static_gates_kernel<<<NWG, 256, 0, stream>>>(
        context, ball, roles, roletab, b0, Wih0p, sg);
    // 3) persistent decoder: all T steps
    lstm_decoder_kernel<<<NWG, 256, SMEM_BYTES, stream>>>(
        out, sg, W_ih0, Whh0b, Wih1b, Whh1b, Wfc1b,
        b1, b_fc1, W_fc2, b_fc2, W_init, b_init, init_pos, maxstep);
}
